// BatchAverage_30253749633130
// MI455X (gfx1250) — compile-verified, matrix operand order checked by reading, not run on hardware
//
#include <hip/hip_runtime.h>
#include <cstdint>

// ---------------------------------------------------------------------------
// Problem constants (from the reference): B=1024 rows, D=128 features,
// NC = B-1 = 1023 gathered columns per row, T = 0.07.
// ---------------------------------------------------------------------------
constexpr int Bsz = 1024;
constexpr int Dd  = 128;
constexpr int NC  = Bsz - 1;

typedef __attribute__((ext_vector_type(16))) _Float16 v16h;
typedef __attribute__((ext_vector_type(8)))  _Float16 v8h;
typedef __attribute__((ext_vector_type(8)))  float    v8f;

// ---------------------------------------------------------------------------
// Stage 1: split f32 features into hi/lo f16 pair (Markidis split).
// feat = hi + lo with hi = fp16(feat), lo = fp16(feat - hi).
// ---------------------------------------------------------------------------
__global__ void __launch_bounds__(256)
split_f32_to_f16(const float* __restrict__ x,
                 _Float16* __restrict__ hi,
                 _Float16* __restrict__ lo, int n) {
  int i = blockIdx.x * blockDim.x + threadIdx.x;
  if (i < n) {
    float v = x[i];
    _Float16 h = (_Float16)v;
    hi[i] = h;
    lo[i] = (_Float16)(v - (float)h);
  }
}

// ---------------------------------------------------------------------------
// Stage 2: Gram matrix G = F F^T via v_wmma_f32_16x16x32_f16, compensated:
//   G ≈ Fhi Fhi^T + Fhi Flo^T + Flo Fhi^T   (f32 accumulate)
//
// One wave owns a 16-row tile x 64-col strip of G (4 accumulators).
// 16-bit A-matrix layout (ISA 7.12.2): lane L in [0,16): row M=L holds
// K = k0 + {0..7} and k0+16+{0..7}; lane L+16 holds K = k0+8+{0..7} and
// k0+24+{0..7}.  B (=rows of F for the column strip, since G is symmetric
// in F) is loaded with the same per-lane K pattern, lane -> column.
// Each fragment = two 16-byte contiguous chunks -> global_load_b128 pairs.
// ---------------------------------------------------------------------------
__device__ __forceinline__ v16h load_frag(const _Float16* __restrict__ base) {
  v8h p0 = *(const v8h*)(base);
  v8h p1 = *(const v8h*)(base + 16);
  v16h f;
#pragma unroll
  for (int e = 0; e < 8; ++e) { f[e] = p0[e]; f[e + 8] = p1[e]; }
  return f;
}

__global__ void __launch_bounds__(256)
gram_wmma(const _Float16* __restrict__ Fh,
          const _Float16* __restrict__ Fl,
          float* __restrict__ G) {
  const int lane = threadIdx.x & 31;
  const int wid  = (blockIdx.x * blockDim.x + threadIdx.x) >> 5;

  const int i0   = (wid >> 4) << 4;   // 64 row tiles of 16
  const int j0   = (wid & 15) << 6;   // 16 col strips of 64
  const int r    = lane & 15;
  const int half = lane >> 4;

  v8f acc[4] = {};

  const int arow = (i0 + r) * Dd;
#pragma unroll
  for (int k0 = 0; k0 < Dd; k0 += 32) {
    const int ka = k0 + half * 8;
    const v16h ah = load_frag(Fh + arow + ka);
    const v16h al = load_frag(Fl + arow + ka);
#pragma unroll
    for (int t = 0; t < 4; ++t) {
      const int brow = (j0 + t * 16 + r) * Dd;
      const v16h bh = load_frag(Fh + brow + ka);
      const v16h bl = load_frag(Fl + brow + ka);
      // hi*hi + hi*lo + lo*hi  (lo*lo term is below f32 rounding)
      acc[t] = __builtin_amdgcn_wmma_f32_16x16x32_f16(
          false, ah, false, bh, (short)0, acc[t], false, false);
      acc[t] = __builtin_amdgcn_wmma_f32_16x16x32_f16(
          false, ah, false, bl, (short)0, acc[t], false, false);
      acc[t] = __builtin_amdgcn_wmma_f32_16x16x32_f16(
          false, al, false, bh, (short)0, acc[t], false, false);
    }
  }

  // C/D layout: VGPR v, lanes 0-15 -> M=v, lanes 16-31 -> M=v+8; N = lane&15.
#pragma unroll
  for (int t = 0; t < 4; ++t) {
    const int col = j0 + t * 16 + r;
#pragma unroll
    for (int v = 0; v < 8; ++v) {
      const int row = i0 + v + half * 8;
      G[row * Bsz + col] = acc[t][v];
    }
  }
}

// ---------------------------------------------------------------------------
// Stage 3: out[b, n] = exp(G[b, idx[b, n]] / T)
// idx reads are fully coalesced; G gathers stay within one 4 KB row (L2-hot).
// ---------------------------------------------------------------------------
__global__ void __launch_bounds__(256)
gather_exp(const float* __restrict__ G,
           const int* __restrict__ idx,
           float* __restrict__ out, int total) {
  int i = blockIdx.x * blockDim.x + threadIdx.x;
  if (i < total) {
    const int b = i / NC;                 // magic-number division
    const int j = idx[i];
    const float s = G[b * Bsz + j];
    out[i] = __expf(s * (1.0f / 0.07f)); // v_exp_f32
  }
}

// ---------------------------------------------------------------------------
// Launch: ws = [ Fhi: 256KB | Flo: 256KB | G: 4MB ]  (all L2-resident)
// ---------------------------------------------------------------------------
extern "C" void kernel_launch(void* const* d_in, const int* in_sizes, int n_in,
                              void* d_out, int out_size, void* d_ws, size_t ws_size,
                              hipStream_t stream) {
  const float* feat = (const float*)d_in[0];
  // d_in[1] (y) is unused by the forward pass.
  const int* idx = (const int*)d_in[2];
  float* out = (float*)d_out;

  char* ws = (char*)d_ws;
  _Float16* Fh = (_Float16*)ws;
  _Float16* Fl = (_Float16*)(ws + (size_t)Bsz * Dd * sizeof(_Float16));
  float*    G  = (float*)   (ws + (size_t)2 * Bsz * Dd * sizeof(_Float16));

  // Stage 1: 131072 elems
  {
    const int n = Bsz * Dd;
    split_f32_to_f16<<<(n + 255) / 256, 256, 0, stream>>>(feat, Fh, Fl, n);
  }
  // Stage 2: (1024/16)*(1024/64) = 1024 waves -> 128 blocks of 8 waves
  {
    const int waves  = (Bsz / 16) * (Bsz / 64);
    const int blocks = waves * 32 / 256;
    gram_wmma<<<blocks, 256, 0, stream>>>(Fh, Fl, G);
  }
  // Stage 3: 1024*1023 elems
  {
    const int total = Bsz * NC;
    gather_exp<<<(total + 255) / 256, 256, 0, stream>>>(G, idx, out, total);
  }
}
